// SelectiveAttentionViT_61289183314185
// MI455X (gfx1250) — compile-verified, matrix operand order checked by reading, not run on hardware
//
#include <hip/hip_runtime.h>
#include <math.h>

typedef __attribute__((ext_vector_type(16))) __bf16 v16bf;
typedef __attribute__((ext_vector_type(8)))  __bf16 v8bf;
typedef __attribute__((ext_vector_type(8)))  float  v8f;

#define DEV __device__ __forceinline__

DEV __bf16 f2bf(float f) {
  unsigned u = __builtin_bit_cast(unsigned, f);
  u += 0x7FFFu + ((u >> 16) & 1u);           // round-to-nearest-even
  unsigned short s = (unsigned short)(u >> 16);
  return __builtin_bit_cast(__bf16, s);
}

DEV v8f vzero8() {
  v8f v;
#pragma unroll
  for (int i = 0; i < 8; ++i) v[i] = 0.f;
  return v;
}

// Fragment (16x32 bf16 A, or [N,K]-row-major B): lane m<16 holds
// K = {ko..ko+7, ko+16..ko+23}; lanes 16..31 the +8 shifted halves
// (caller passes ko = k0 + (lane>>4)*8).
DEV v16bf ld_frag_bf(const __bf16* p) {
  union { v16bf v; v8bf h[2]; } u;
  u.h[0] = *(const v8bf*)(p);
  u.h[1] = *(const v8bf*)(p + 16);
  return u.v;
}

DEV v8f wmma_bf16(v16bf a, v16bf b, v8f c) {
  return __builtin_amdgcn_wmma_f32_16x16x32_bf16(false, a, false, b, (short)0, c,
                                                 false, false);
}

// ---------------- wave-tiled bf16 WMMA GEMM: C = A[M,K] * W[N,K]^T ----------
// Each wave owns a 32x64 output tile: 2 A-fragments x 4 B-fragments,
// 8 WMMAs per 32-wide K step. Wave/tile indices are pinned to SGPRs via
// readfirstlane so all tile-validity tests are scalar branches (EXEC stays
// all-ones through every WMMA). Epilogue functor gets (batch,row,col,val).
template <class Epi>
__global__ __launch_bounds__(256) void gemm_k(
    const __bf16* __restrict__ A, const __bf16* __restrict__ W,
    int M, int N, int K, int lda, int ldw,
    long aBatch, long wBatch, Epi epi)
{
  const int batch = blockIdx.y;
  A += (long)batch * aBatch;
  W += (long)batch * wBatch;
  const int tilesM  = (M + 31) >> 5;
  const int tilesNG = (N + 63) >> 6;
  // wave index as a compiler-provable scalar (lane0's threadIdx.x >> 5)
  const int wvid = __builtin_amdgcn_readfirstlane(threadIdx.x) >> 5;
  const int wave = blockIdx.x * 8 + wvid;
  if (wave >= tilesM * tilesNG) return;
  const int mi = wave % tilesM;
  const int ng = wave / tilesM;
  const int lane = threadIdx.x & 31;
  const int r  = lane & 15;
  const int hi = lane >> 4;

  const __bf16* ap[2];
#pragma unroll
  for (int t = 0; t < 2; ++t) {
    int arow = mi * 32 + t * 16 + r; if (arow > M - 1) arow = M - 1;
    ap[t] = A + (long)arow * lda + hi * 8;
  }

  const __bf16* wp[4];
#pragma unroll
  for (int j = 0; j < 4; ++j) {
    int wrow = ng * 64 + j * 16 + r; if (wrow > N - 1) wrow = N - 1;
    wp[j] = W + (long)wrow * ldw + hi * 8;
  }

  v8f acc[2][4];
#pragma unroll
  for (int t = 0; t < 2; ++t)
#pragma unroll
    for (int j = 0; j < 4; ++j) acc[t][j] = vzero8();

  if ((ng + 1) * 64 <= N) {
    // ---- fast path: all four N-subtiles valid, branch-free inner loop ----
#pragma unroll 2
    for (int k0 = 0; k0 < K; k0 += 32) {
      v16bf a0 = ld_frag_bf(ap[0] + k0);
      v16bf a1 = ld_frag_bf(ap[1] + k0);
      v16bf b0 = ld_frag_bf(wp[0] + k0);
      v16bf b1 = ld_frag_bf(wp[1] + k0);
      v16bf b2 = ld_frag_bf(wp[2] + k0);
      v16bf b3 = ld_frag_bf(wp[3] + k0);
      acc[0][0] = wmma_bf16(a0, b0, acc[0][0]);
      acc[1][0] = wmma_bf16(a1, b0, acc[1][0]);
      acc[0][1] = wmma_bf16(a0, b1, acc[0][1]);
      acc[1][1] = wmma_bf16(a1, b1, acc[1][1]);
      acc[0][2] = wmma_bf16(a0, b2, acc[0][2]);
      acc[1][2] = wmma_bf16(a1, b2, acc[1][2]);
      acc[0][3] = wmma_bf16(a0, b3, acc[0][3]);
      acc[1][3] = wmma_bf16(a1, b3, acc[1][3]);
    }
  } else {
    // ---- edge path: scalar-guarded subtiles ----
    bool nv1 = (ng * 64 + 16 < N), nv2 = (ng * 64 + 32 < N), nv3 = (ng * 64 + 48 < N);
    for (int k0 = 0; k0 < K; k0 += 32) {
      v16bf a0 = ld_frag_bf(ap[0] + k0);
      v16bf a1 = ld_frag_bf(ap[1] + k0);
      v16bf b0 = ld_frag_bf(wp[0] + k0);
      acc[0][0] = wmma_bf16(a0, b0, acc[0][0]);
      acc[1][0] = wmma_bf16(a1, b0, acc[1][0]);
      if (nv1) {
        v16bf b = ld_frag_bf(wp[1] + k0);
        acc[0][1] = wmma_bf16(a0, b, acc[0][1]);
        acc[1][1] = wmma_bf16(a1, b, acc[1][1]);
      }
      if (nv2) {
        v16bf b = ld_frag_bf(wp[2] + k0);
        acc[0][2] = wmma_bf16(a0, b, acc[0][2]);
        acc[1][2] = wmma_bf16(a1, b, acc[1][2]);
      }
      if (nv3) {
        v16bf b = ld_frag_bf(wp[3] + k0);
        acc[0][3] = wmma_bf16(a0, b, acc[0][3]);
        acc[1][3] = wmma_bf16(a1, b, acc[1][3]);
      }
    }
  }

#pragma unroll
  for (int t = 0; t < 2; ++t) {
#pragma unroll
    for (int j = 0; j < 4; ++j) {
      if (ng * 64 + j * 16 >= N) continue;          // scalar test
      const int col = ng * 64 + j * 16 + r;
      if (col >= N) continue;                        // lane edge test
#pragma unroll
      for (int e = 0; e < 8; ++e) {
        const int row = mi * 32 + t * 16 + hi * 8 + e;
        if (row < M) epi(batch, row, col, acc[t][j][e]);
      }
    }
  }
}

// ---------------- epilogues ------------------------------------------------
struct PatchEmbedEpi {                // rows are patches; write into h[+1..]
  float* h; const float* pb; const float* pos;
  DEV void operator()(int, int row, int col, float v) const {
    const int b = row / 196, t = row % 196 + 1;
    h[((long)(b * 197 + t)) * 768 + col] = v + pb[col] + pos[t * 768 + col];
  }
};
struct QkvEpi {                       // scatter q(scaled)/k/v^T, padded N=224
  __bf16* q; __bf16* k; __bf16* vt; float scale;
  DEV void operator()(int, int row, int col, float v) const {
    const int b = row / 197, n = row % 197;
    const int sec = col / 768, rem = col % 768, hh = rem >> 6, d = rem & 63;
    const long bh = b * 12 + hh;
    if (sec == 0)      q[(bh * 224 + n) * 64 + d] = f2bf(v * scale);
    else if (sec == 1) k[(bh * 224 + n) * 64 + d] = f2bf(v);
    else               vt[(bh * 64 + d) * 224 + n] = f2bf(v);
  }
};
struct LogitsEpi {
  float* lg;
  DEV void operator()(int bh, int row, int col, float v) const {
    lg[((long)bh * 197 + row) * 224 + col] = v;
  }
};
struct AVEpi {
  __bf16* o;
  DEV void operator()(int bh, int row, int col, float v) const {
    const int b = bh / 12, hh = bh % 12;
    o[((long)(b * 197 + row)) * 768 + hh * 64 + col] = f2bf(v);
  }
};
struct ResidEpi {                     // h += val + bias   (proj / mlp2)
  float* h; const float* bias;
  DEV void operator()(int, int row, int col, float v) const {
    h[(long)row * 768 + col] += v + bias[col];
  }
};
struct GeluEpi {                      // exact-erf GELU, bf16 out (mlp hidden)
  __bf16* out; const float* bias;
  DEV void operator()(int, int row, int col, float v) const {
    const float x = v + bias[col];
    const float g = 0.5f * x * (1.f + erff(x * 0.70710678118654752f));
    out[(long)row * 3072 + col] = f2bf(g);
  }
};
struct HeadEpi {
  float* out; const float* bias;
  DEV void operator()(int, int row, int col, float v) const {
    out[row * 1000 + col] = v + bias[col];
  }
};

// ---------------- bulk f32 -> bf16 weight conversion -----------------------
__global__ __launch_bounds__(256) void cvtw_k(const float* __restrict__ in,
                                              __bf16* __restrict__ out, long n)
{
  const long i = ((long)blockIdx.x * blockDim.x + threadIdx.x) * 8;
  if (i >= n) return;
  v8f x = *(const v8f*)(in + i);
  v8bf y;
#pragma unroll
  for (int t = 0; t < 8; ++t) y[t] = f2bf(x[t]);
  *(v8bf*)(out + i) = y;
}

// ---------------- LayerNorm (one wave per 768-wide row) -> bf16 ------------
__global__ __launch_bounds__(256) void ln_k(
    const float* __restrict__ x, long xRowStride,
    const float* __restrict__ w, const float* __restrict__ b,
    __bf16* __restrict__ out, int rows)
{
  const int wave = blockIdx.x * (blockDim.x >> 5) + (threadIdx.x >> 5);
  if (wave >= rows) return;
  const int lane = threadIdx.x & 31;
  const float* xp = x + (long)wave * xRowStride;
  float v[24];
  float s = 0.f;
#pragma unroll
  for (int j = 0; j < 24; ++j) { v[j] = xp[lane + j * 32]; s += v[j]; }
#pragma unroll
  for (int off = 16; off > 0; off >>= 1) s += __shfl_xor(s, off, 32);
  const float mu = s * (1.f / 768.f);
  float q = 0.f;
#pragma unroll
  for (int j = 0; j < 24; ++j) { const float d = v[j] - mu; q += d * d; }
#pragma unroll
  for (int off = 16; off > 0; off >>= 1) q += __shfl_xor(q, off, 32);
  const float rs = rsqrtf(q * (1.f / 768.f) + 1e-5f);
  __bf16* op = out + (long)wave * 768;
#pragma unroll
  for (int j = 0; j < 24; ++j) {
    const int c = lane + j * 32;
    op[c] = f2bf((v[j] - mu) * rs * w[c] + b[c]);
  }
}

// ---------------- attention row normalize (softmax / L2Q polynomial) -------
__global__ __launch_bounds__(256) void attn_norm_k(
    const float* __restrict__ lg, __bf16* __restrict__ attn,
    const float* __restrict__ alpha, const float* __restrict__ beta,
    const float* __restrict__ gamma, int poly)
{
  const int wave = blockIdx.x * (blockDim.x >> 5) + (threadIdx.x >> 5);
  if (wave >= 384 * 197) return;
  const int bh = wave / 197, row = wave % 197;
  const int hh = bh % 12;
  const int lane = threadIdx.x & 31;
  const float* rp = lg + ((long)bh * 197 + row) * 224;
  float v[7], o[7];
#pragma unroll
  for (int j = 0; j < 7; ++j) { const int c = lane + j * 32; v[j] = (c < 197) ? rp[c] : 0.f; }
  if (poly) {
    const float a = alpha[hh], bb = beta[hh], g = gamma[hh];
    float s = 0.f;
#pragma unroll
    for (int j = 0; j < 7; ++j) {
      const int c = lane + j * 32;
      const float t = (c < 197) ? fmaxf(a * v[j] * v[j] + bb * v[j] + g, 0.f) : 0.f;
      o[j] = t; s += t;
    }
#pragma unroll
    for (int off = 16; off > 0; off >>= 1) s += __shfl_xor(s, off, 32);
    const float inv = 1.f / (s + 1e-6f);
#pragma unroll
    for (int j = 0; j < 7; ++j) o[j] *= inv;
  } else {
    float m = -3.4e38f;
#pragma unroll
    for (int j = 0; j < 7; ++j) { const int c = lane + j * 32; if (c < 197 && v[j] > m) m = v[j]; }
#pragma unroll
    for (int off = 16; off > 0; off >>= 1) m = fmaxf(m, __shfl_xor(m, off, 32));
    float s = 0.f;
#pragma unroll
    for (int j = 0; j < 7; ++j) {
      const int c = lane + j * 32;
      const float t = (c < 197) ? expf(v[j] - m) : 0.f;
      o[j] = t; s += t;
    }
#pragma unroll
    for (int off = 16; off > 0; off >>= 1) s += __shfl_xor(s, off, 32);
    const float inv = 1.f / s;
#pragma unroll
    for (int j = 0; j < 7; ++j) o[j] *= inv;
  }
  __bf16* op = attn + ((long)bh * 197 + row) * 224;
#pragma unroll
  for (int j = 0; j < 7; ++j) { const int c = lane + j * 32; op[c] = f2bf(o[j]); }
}

// ---------------- patchify: x[B,3,224,224] f32 -> bf16 im2col [6272,768] ---
__global__ void patchify_k(const float* __restrict__ x, __bf16* __restrict__ out)
{
  const long idx = (long)blockIdx.x * blockDim.x + threadIdx.x;
  if (idx >= (long)32 * 196 * 768) return;
  const int e = (int)(idx % 768);
  const int p = (int)((idx / 768) % 196);
  const int b = (int)(idx / (768 * 196));
  const int c = e >> 8, rr = (e >> 4) & 15, ss = e & 15;
  const int gy = p / 14, gx = p % 14;
  out[idx] = f2bf(x[(((long)b * 3 + c) * 224 + gy * 16 + rr) * 224 + gx * 16 + ss]);
}

__global__ void clspos_k(const float* __restrict__ cls, const float* __restrict__ pos,
                         float* __restrict__ h)
{
  const int idx = blockIdx.x * blockDim.x + threadIdx.x;
  if (idx >= 32 * 768) return;
  const int b = idx / 768, d = idx % 768;
  h[((long)b * 197) * 768 + d] = cls[d] + pos[d];
}

// ---------------- host-side helpers ----------------------------------------
template <class Epi>
static void launch_gemm(const __bf16* A, const __bf16* W, int M, int N, int K,
                        int lda, int ldw, long aBatch, long wBatch, int batches,
                        Epi epi, hipStream_t s)
{
  const int tilesM = (M + 31) / 32, tilesNG = (N + 63) / 64;
  const int waves = tilesM * tilesNG;
  dim3 grid((waves + 7) / 8, batches);
  gemm_k<Epi><<<grid, 256, 0, s>>>(A, W, M, N, K, lda, ldw, aBatch, wBatch, epi);
}

static void launch_cvt(const float* in, __bf16* out, long n, hipStream_t s)
{
  const long thr = n / 8;                 // n is always a multiple of 8
  cvtw_k<<<(unsigned)((thr + 255) / 256), 256, 0, s>>>(in, out, n);
}

extern "C" void kernel_launch(void* const* d_in, const int* in_sizes, int n_in,
                              void* d_out, int out_size, void* d_ws, size_t ws_size,
                              hipStream_t stream)
{
  (void)in_sizes; (void)n_in; (void)out_size; (void)ws_size;
  const float* x       = (const float*)d_in[0];
  const float* patch_w = (const float*)d_in[1];
  const float* patch_b = (const float*)d_in[2];
  const float* cls_tok = (const float*)d_in[3];
  const float* pos     = (const float*)d_in[4];
  const float* ln1_w   = (const float*)d_in[5];
  const float* ln1_b   = (const float*)d_in[6];
  const float* qkv_w   = (const float*)d_in[7];
  const float* proj_w  = (const float*)d_in[8];
  const float* proj_b  = (const float*)d_in[9];
  const float* ln2_w   = (const float*)d_in[10];
  const float* ln2_b   = (const float*)d_in[11];
  const float* mlp_w1  = (const float*)d_in[12];
  const float* mlp_b1  = (const float*)d_in[13];
  const float* mlp_w2  = (const float*)d_in[14];
  const float* mlp_b2  = (const float*)d_in[15];
  const float* alpha   = (const float*)d_in[16];
  const float* beta    = (const float*)d_in[17];
  const float* gamma   = (const float*)d_in[18];
  const float* norm_w  = (const float*)d_in[19];
  const float* norm_b  = (const float*)d_in[20];
  const float* head_w  = (const float*)d_in[21];
  const float* head_b  = (const float*)d_in[22];

  size_t off = 0;
  auto carve = [&](size_t bytes) -> void* {
    off = (off + 255) & ~(size_t)255;
    void* p = (char*)d_ws + off;
    off += bytes;
    return p;
  };
  // activations
  float*  h       = (float*) carve((size_t)6304 * 768 * 4);   // tokens, f32
  __bf16* patches = (__bf16*)carve((size_t)6272 * 768 * 2);   // im2col
  __bf16* xln     = (__bf16*)carve((size_t)6304 * 768 * 2);   // LN output
  __bf16* obf     = (__bf16*)carve((size_t)6304 * 768 * 2);   // attn output
  __bf16* mbf     = (__bf16*)carve((size_t)6304 * 3072 * 2);  // MLP hidden
  __bf16* qb      = (__bf16*)carve((size_t)384 * 224 * 64 * 2);
  __bf16* kb      = (__bf16*)carve((size_t)384 * 224 * 64 * 2);
  __bf16* vtb     = (__bf16*)carve((size_t)384 * 64 * 224 * 2);
  float*  lgt     = (float*) carve((size_t)384 * 197 * 224 * 4);
  __bf16* attn    = (__bf16*)carve((size_t)384 * 197 * 224 * 2);
  __bf16* clsbf   = (__bf16*)carve((size_t)32 * 768 * 2);
  // bf16 weight mirrors (converted once per call)
  const long nPW = 768L * 768, nQW = 12L * 2304 * 768, nPR = 12L * 768 * 768;
  const long nM1 = 12L * 3072 * 768, nM2 = 12L * 768 * 3072, nHW = 1000L * 768;
  __bf16* pwb  = (__bf16*)carve((size_t)nPW * 2);
  __bf16* qwb  = (__bf16*)carve((size_t)nQW * 2);
  __bf16* prwb = (__bf16*)carve((size_t)nPR * 2);
  __bf16* m1wb = (__bf16*)carve((size_t)nM1 * 2);
  __bf16* m2wb = (__bf16*)carve((size_t)nM2 * 2);
  __bf16* hwb  = (__bf16*)carve((size_t)nHW * 2);

  // ---- one-shot weight conversion (L2-amortized) ----
  launch_cvt(patch_w, pwb,  nPW, stream);
  launch_cvt(qkv_w,   qwb,  nQW, stream);
  launch_cvt(proj_w,  prwb, nPR, stream);
  launch_cvt(mlp_w1,  m1wb, nM1, stream);
  launch_cvt(mlp_w2,  m2wb, nM2, stream);
  launch_cvt(head_w,  hwb,  nHW, stream);

  // ---- patch embedding ----
  {
    const long total = (long)32 * 196 * 768;
    patchify_k<<<(unsigned)((total + 255) / 256), 256, 0, stream>>>(x, patches);
  }
  launch_gemm(patches, pwb, 6272, 768, 768, 768, 768, 0L, 0L, 1,
              PatchEmbedEpi{h, patch_b, pos}, stream);
  clspos_k<<<(32 * 768 + 255) / 256, 256, 0, stream>>>(cls_tok, pos, h);

  // ---- transformer layers ----
  for (int i = 0; i < 12; ++i) {
    ln_k<<<(6304 + 7) / 8, 256, 0, stream>>>(h, 768L, ln1_w + i * 768, ln1_b + i * 768, xln, 6304);
    launch_gemm(xln, qwb + (size_t)i * 2304 * 768, 6304, 2304, 768, 768, 768, 0L, 0L, 1,
                QkvEpi{qb, kb, vtb, 0.125f}, stream);
    // logits = (q*scale) . k^T, batched over 384 (b,h) pairs
    launch_gemm(qb, kb, 197, 197, 64, 64, 64, 224L * 64, 224L * 64, 384,
                LogitsEpi{lgt}, stream);
    attn_norm_k<<<(384 * 197 + 7) / 8, 256, 0, stream>>>(
        lgt, attn, alpha + i * 12, beta + i * 12, gamma + i * 12, (i % 2 == 0) ? 1 : 0);
    // o = attn . v   (W = v^T stored [64,224] row-major; padded K cols hit zero attn)
    launch_gemm(attn, vtb, 197, 64, 224, 224, 224, 197L * 224, 64L * 224, 384,
                AVEpi{obf}, stream);
    launch_gemm(obf, prwb + (size_t)i * 768 * 768, 6304, 768, 768, 768, 768, 0L, 0L, 1,
                ResidEpi{h, proj_b + i * 768}, stream);
    ln_k<<<(6304 + 7) / 8, 256, 0, stream>>>(h, 768L, ln2_w + i * 768, ln2_b + i * 768, xln, 6304);
    launch_gemm(xln, m1wb + (size_t)i * 3072 * 768, 6304, 3072, 768, 768, 768, 0L, 0L, 1,
                GeluEpi{mbf, mlp_b1 + i * 3072}, stream);
    launch_gemm(mbf, m2wb + (size_t)i * 768 * 3072, 6304, 768, 3072, 3072, 3072, 0L, 0L, 1,
                ResidEpi{h, mlp_b2 + i * 768}, stream);
  }

  // ---- head ----
  ln_k<<<(32 + 7) / 8, 256, 0, stream>>>(h, 197L * 768, norm_w, norm_b, clsbf, 32);
  launch_gemm(clsbf, hwb, 32, 1000, 768, 768, 768, 0L, 0L, 1,
              HeadEpi{(float*)d_out, head_b}, stream);
}
